// CRF_9844065042750
// MI455X (gfx1250) — matrix operand order repaired and checked by reading, not compiled, not run on hardware
//
#include <hip/hip_runtime.h>

// CRF NLL for MI455X (gfx1250, wave32).
// Forward recursion as per-step exp-domain GEMM on v_wmma_f32_16x16x32_f16,
// transposed:  u = E * Q,  E = exp(trans) constant in A-operand VGPR fragments.
// State kept in SCALED-EXP domain (q = exp(score - M), running per-batch log-scale M):
//   per step only 1 v_log + 1 v_rcp on the recursion path; the 24 emission exps
//   depend only on prefetched feats (off the critical chain).
// Cross-lane data movement: v_permlanex16 only (no LDS / DS counters in hot loop).

typedef __attribute__((ext_vector_type(16))) _Float16 v16h;
typedef __attribute__((ext_vector_type(8)))  _Float16 v8h;
typedef __attribute__((ext_vector_type(8)))  float    v8f;
typedef __attribute__((ext_vector_type(4)))  float    v4f;

#define NTAGS    48
#define KPAD     64
#define SLEN     1024
#define NBATCH   512
#define START_T  45
#define STOP_T   46
#define NEGV     (-10000.0f)
#define NWAVES   4
#define NROWS    16

#define L2E 1.4426950408889634f   // log2(e)
#define LN2 0.6931471805599453f   // ln(2)

__device__ __forceinline__ v16h pack16(v8h lo, v8h hi) {
  v16h r;
#pragma unroll
  for (int i = 0; i < 8; ++i) { r[i] = lo[i]; r[i + 8] = hi[i]; }
  return r;
}

// exchange with lane^16 (identity cross-half swap) -- pure VALU, no DS
__device__ __forceinline__ float xhalf(float x) {
  const int xi = __builtin_bit_cast(int, x);
  const int r  = __builtin_amdgcn_permlanex16(xi, xi, 0x76543210, 0xfedcba98, false, false);
  return __builtin_bit_cast(float, r);
}

__global__ __launch_bounds__(32 * NWAVES) void crf_forward_kernel(
    const float* __restrict__ feats, const float* __restrict__ masks,
    const float* __restrict__ trans, const float* __restrict__ gold,
    float* __restrict__ out)
{
  __shared__ __align__(16) _Float16 eBuf[NTAGS][KPAD];   // exp(trans[j][i]), i zero-padded
  __shared__ float stopRow[NTAGS];

  const int tid  = threadIdx.x;
  const int wv   = tid >> 5;
  const int lane = tid & 31;
  const int hf   = lane >> 4;
  const int nl   = lane & 15;

  // ---- one-time init (only LDS use in the kernel)
  for (int idx = tid; idx < NTAGS * KPAD; idx += 32 * NWAVES) {
    const int j = idx >> 6, i = idx & 63;
    eBuf[j][i] = (_Float16)((i < NTAGS) ? __expf(trans[j * NTAGS + i]) : 0.0f);
  }
  for (int j = tid; j < NTAGS; j += 32 * NWAVES)
    stopRow[j] = trans[STOP_T * NTAGS + j];
  __syncthreads();

  // ---- constant A fragments of E (contiguous k order: element s holds k = 32kc + 16hf + s, same as B)
  v16h afrag[3][2];
#pragma unroll
  for (int mt = 0; mt < 3; ++mt)
#pragma unroll
    for (int kc = 0; kc < 2; ++kc) {
      const v8h* lo = (const v8h*)&eBuf[nl + 16 * mt][kc * 32 + 16 * hf];
      const v8h* hi = (const v8h*)&eBuf[nl + 16 * mt][kc * 32 + 16 * hf + 8];
      afrag[mt][kc] = pack16(lo[0], hi[0]);
    }

  const int b0  = (blockIdx.x * NWAVES + wv) * NROWS;
  const int myb = b0 + nl;
  const float* fptr = feats + (size_t)myb * SLEN * NTAGS + 8 * hf;  // +NTAGS per step
  const float* mptr = masks + (size_t)myb * SLEN;

  // scaled-exp state: q[n][r] = exp(score_T[tag = 16n+8hf+r][myb] - M)
  float q[3][8];
  float M = 0.0f;
#pragma unroll
  for (int n = 0; n < 3; ++n)
#pragma unroll
    for (int r = 0; r < 8; ++r)
      q[n][r] = ((16 * n + 8 * hf + r) == START_T) ? 1.0f : 0.0f;  // exp(0) / exp(-1e4)

  // one recursion step: emissions ecur + mask mkcur -> updates (q, M)
  auto step = [&](const float (&ecur)[3][8], float mkcur) {
    // emission exponentials -- independent of state, off the critical chain
    float ee[3][8];
#pragma unroll
    for (int n = 0; n < 3; ++n)
#pragma unroll
      for (int r = 0; r < 8; ++r)
        ee[n][r] = __builtin_amdgcn_exp2f(ecur[n][r] * L2E);

    // B fragments of Q: element h of chunk kc holds tag i = 32kc + 16hf + h
    float xq[3][8];
#pragma unroll
    for (int n = 0; n < 3; ++n)
#pragma unroll
      for (int r = 0; r < 8; ++r) xq[n][r] = xhalf(q[n][r]);

    v16h bf0, bf1;
#pragma unroll
    for (int r = 0; r < 8; ++r) {
      bf0[r]     = (_Float16)(hf ? xq[1][r] : q[0][r]);   // tag 16hf + r
      bf0[r + 8] = (_Float16)(hf ? q[1][r]  : xq[0][r]);  // tag 16hf + 8 + r
      bf1[r]     = (_Float16)(hf ? 0.0f     : q[2][r]);   // tag 32 + 16hf + r (>=48 -> 0)
      bf1[r + 8] = (_Float16)(hf ? 0.0f     : xq[2][r]);  // tag 40 + 16hf + r (>=48 -> 0)
    }

    // acc = E * Q   (3 j-tiles x 2 K-chunks)
    const v8f zero = {};
    v8f accv[3];
#pragma unroll
    for (int mt = 0; mt < 3; ++mt) {
      v8f c = __builtin_amdgcn_wmma_f32_16x16x32_f16(false, afrag[mt][0], false, bf0,
                                                     (short)0, zero, false, false);
      accv[mt] = __builtin_amdgcn_wmma_f32_16x16x32_f16(false, afrag[mt][1], false, bf1,
                                                        (short)0, c, false, false);
    }

    // u = acc * exp(emit);  c = max(u) per batch column (local 24 + one exchange)
    float u[3][8];
    float cm = 0.0f;                       // u >= 0
#pragma unroll
    for (int n = 0; n < 3; ++n)
#pragma unroll
      for (int r = 0; r < 8; ++r) {
        u[n][r] = accv[n][r] * ee[n][r];
        cm = fmaxf(cm, u[n][r]);
      }
    cm = fmaxf(cm, xhalf(cm));
    const float c    = cm + 1e-30f;        // additive guard: no clamp/canonicalize
    const float rc   = __builtin_amdgcn_rcpf(c);
    const float dM   = LN2 * __builtin_amdgcn_logf(c);
    const float mkrc = mkcur * rc;
    const float omk  = 1.0f - mkcur;

    // exact reference blend: new*m + old*(1-m)
#pragma unroll
    for (int n = 0; n < 3; ++n)
#pragma unroll
      for (int r = 0; r < 8; ++r)
        q[n][r] = fmaf(mkrc, u[n][r], omk * q[n][r]);
    M = fmaf(mkcur, dM, M);
  };

  // emissions/mask register double-buffer
  float ec[3][8], en[3][8], mkc, mkn;
#pragma unroll
  for (int n = 0; n < 3; ++n) {
    const v4f u = *(const v4f*)(fptr + 16 * n);
    const v4f v = *(const v4f*)(fptr + 16 * n + 4);
#pragma unroll
    for (int qq = 0; qq < 4; ++qq) { ec[n][qq] = u[qq]; ec[n][qq + 4] = v[qq]; }
  }
  mkc = mptr[0];

#pragma unroll 2
  for (int t = 0; t < SLEN - 1; ++t) {
    const float* fnx = fptr + NTAGS;
#pragma unroll
    for (int n = 0; n < 3; ++n) {
      const v4f u = *(const v4f*)(fnx + 16 * n);
      const v4f v = *(const v4f*)(fnx + 16 * n + 4);
#pragma unroll
      for (int qq = 0; qq < 4; ++qq) { en[n][qq] = u[qq]; en[n][qq + 4] = v[qq]; }
    }
    mkn = mptr[t + 1];
    __builtin_prefetch(fptr + 8 * NTAGS, 0, 1);   // fixed immediate offset, non-faulting

    step(ec, mkc);

#pragma unroll
    for (int n = 0; n < 3; ++n)
#pragma unroll
      for (int r = 0; r < 8; ++r) ec[n][r] = en[n][r];
    mkc = mkn;
    fptr = fnx;
  }
  step(ec, mkc);   // peeled final step (t = SLEN-1)

  // ---- logZ = M + log( sum_tags q * exp(trans[STOP]) );  out = logZ - gold
  float sm = 0.0f;
#pragma unroll
  for (int n = 0; n < 3; ++n)
#pragma unroll
    for (int r = 0; r < 8; ++r)
      sm += q[n][r] * __expf(stopRow[16 * n + 8 * hf + r]);
  sm += xhalf(sm);
  if (hf == 0)
    out[myb] = M + __logf(sm) - gold[myb];
}

// ---- gold path score (parallel gather + block reduction) ----
__global__ __launch_bounds__(256) void crf_gold_kernel(
    const float* __restrict__ feats, const float* __restrict__ masks,
    const int* __restrict__ tags, const float* __restrict__ trans,
    float* __restrict__ gold)
{
  __shared__ float redA[256], redM[256];
  const int b = blockIdx.x, tid = threadIdx.x;
  float acc = 0.0f, mc = 0.0f;
  for (int t = tid; t < SLEN; t += 256) {
    const float mk = masks[(size_t)b * SLEN + t];
    const int tg = tags[(size_t)b * SLEN + t];
    const int pv = (t == 0) ? START_T : tags[(size_t)b * SLEN + t - 1];
    acc += (feats[((size_t)b * SLEN + t) * NTAGS + tg] + trans[tg * NTAGS + pv]) * mk;
    mc  += mk;
  }
  redA[tid] = acc; redM[tid] = mc;
  __syncthreads();
  for (int stp = 128; stp > 0; stp >>= 1) {
    if (tid < stp) { redA[tid] += redA[tid + stp]; redM[tid] += redM[tid + stp]; }
    __syncthreads();
  }
  if (tid == 0) {
    const int last = (int)(redM[0] + 0.5f);
    const int lastTag = (last == 0) ? START_T : tags[(size_t)b * SLEN + last - 1];
    gold[b] = redA[0] + trans[STOP_T * NTAGS + lastTag];
  }
}

extern "C" void kernel_launch(void* const* d_in, const int* in_sizes, int n_in,
                              void* d_out, int out_size, void* d_ws, size_t ws_size,
                              hipStream_t stream) {
  const float* feats = (const float*)d_in[0];
  const float* masks = (const float*)d_in[1];
  const int*   tags  = (const int*)d_in[2];
  const float* trans = (const float*)d_in[3];
  float* out  = (float*)d_out;
  float* gold = (float*)d_ws;

  crf_gold_kernel<<<NBATCH, 256, 0, stream>>>(feats, masks, tags, trans, gold);
  crf_forward_kernel<<<NBATCH / (NWAVES * NROWS), 32 * NWAVES, 0, stream>>>(
      feats, masks, trans, gold, out);
}
